// SlidingWindowAttention_61383672594956
// MI455X (gfx1250) — compile-verified, matrix operand order checked by reading, not run on hardware
//
#include <hip/hip_runtime.h>
#include <math.h>

typedef __attribute__((ext_vector_type(16))) _Float16 v16h;
typedef __attribute__((ext_vector_type(4)))  _Float16 v4h;
typedef __attribute__((ext_vector_type(8)))  float    v8f;

#define S_LEN   4096
#define DHEAD   64
#define WINLEN  512
#define HALFW   256
#define NWIN    16              // 15 full windows + 1 tail
#define VT_PITCH 528            // 512 + 16 halfs pad: keeps 32B alignment, spreads banks
#define THREADS 512             // 16 wave32 waves
#define SMEM_HALFS (WINLEN*DHEAD + DHEAD*VT_PITCH + 16*512)  // Ksh + Vt + per-wave P

__device__ __forceinline__ v8f v8f_zero() {
    v8f z;
#pragma unroll
    for (int i = 0; i < 8; ++i) z[i] = 0.0f;
    return z;
}

__global__ void __launch_bounds__(THREADS, 1)
swa_wmma_kernel(const float* __restrict__ Qg, const float* __restrict__ Kg,
                const float* __restrict__ Vg, const float* __restrict__ scale_p,
                float* __restrict__ outg)
{
    extern __shared__ _Float16 smem[];
    _Float16* Ksh = smem;                       // [len][64] f16 row-major
    _Float16* Vt  = smem + WINLEN * DHEAD;      // [64][VT_PITCH] f16 (V transposed)
    _Float16* Pb  = Vt + DHEAD * VT_PITCH;      // 16 waves x (16x32) f16 score tiles

    const int w     = blockIdx.x & (NWIN - 1);  // window index 0..15
    const int bh    = blockIdx.x >> 4;          // (b,h) flat 0..31
    const int start = HALFW * w;                // window start row (tail: 3840)
    const int len   = (w < NWIN - 1) ? WINLEN : HALFW;

    const size_t base = ((size_t)bh * S_LEN + start) * DHEAD;
    const float* Qb = Qg + base;
    const float* Kb = Kg + base;
    const float* Vb = Vg + base;
    float* outB = outg + ((size_t)bh * S_LEN + start) * DHEAD;

    const float sc = scale_p[0];
    const int tid = threadIdx.x;

    // ---- stage K (f16, [key][d]) and V^T (f16, [d][key]) into LDS ----
    const int total4 = (len * DHEAD) >> 2;      // float4 elements
    const float4* K4 = (const float4*)Kb;
    const float4* V4 = (const float4*)Vb;
    for (int i = tid; i < total4; i += THREADS) {
        float4 kv = K4[i];
        v4h h;
        h[0] = (_Float16)kv.x; h[1] = (_Float16)kv.y;
        h[2] = (_Float16)kv.z; h[3] = (_Float16)kv.w;
        *(v4h*)(Ksh + i * 4) = h;

        float4 vv = V4[i];
        int key = i >> 4;            // i*4 / 64
        int d0  = (i & 15) << 2;     // (i*4) % 64
        Vt[(d0 + 0) * VT_PITCH + key] = (_Float16)vv.x;
        Vt[(d0 + 1) * VT_PITCH + key] = (_Float16)vv.y;
        Vt[(d0 + 2) * VT_PITCH + key] = (_Float16)vv.z;
        Vt[(d0 + 3) * VT_PITCH + key] = (_Float16)vv.w;
    }
    __syncthreads();

    const int wave = tid >> 5;
    const int lane = tid & 31;
    const int lrow = lane & 15;      // row/col within 16-tile
    const int lhi  = lane >> 4;      // k-chunk half select
    const int qwb  = wave * 32;      // this wave's query-row base (2 q-tiles)
    if (qwb >= len) return;          // tail window: waves 8..15 done after staging

    _Float16* Pw = Pb + wave * 512;  // private 16x32 f16 scratch

    // ---- Q A-tiles in registers, pre-scaled by 1/sqrt(D) ----
    v16h qa[2][2];
#pragma unroll
    for (int qt = 0; qt < 2; ++qt) {
        const float* qrow = Qb + (size_t)(qwb + qt * 16 + lrow) * DHEAD;
#pragma unroll
        for (int t = 0; t < 2; ++t) {
            const float* qp = qrow + t * 32 + lhi * 16;
            v16h a;
#pragma unroll
            for (int e = 0; e < 16; ++e) a[e] = (_Float16)(qp[e] * sc);
            qa[qt][t] = a;
        }
    }

    v8f o[2][4];
    v8f dsum[2];
#pragma unroll
    for (int qt = 0; qt < 2; ++qt) {
        dsum[qt] = v8f_zero();
#pragma unroll
        for (int nt = 0; nt < 4; ++nt) o[qt][nt] = v8f_zero();
    }

    // ---- stream keys in 32-wide chunks: scores -> sigmoid -> P -> P*V ----
    const int nkc = len >> 5;
    for (int kc = 0; kc < nkc; ++kc) {
        const _Float16* krL = Ksh + (kc * 32 + lrow) * DHEAD + lhi * 16;
        const _Float16* krR = krL + 16 * DHEAD;
        v16h bkL0 = *(const v16h*)(krL);
        v16h bkL1 = *(const v16h*)(krL + 32);
        v16h bkR0 = *(const v16h*)(krR);
        v16h bkR1 = *(const v16h*)(krR + 32);
        const int kb = kc * 32 + lhi * 16;
        v16h bv0 = *(const v16h*)(Vt + ( 0 + lrow) * VT_PITCH + kb);
        v16h bv1 = *(const v16h*)(Vt + (16 + lrow) * VT_PITCH + kb);
        v16h bv2 = *(const v16h*)(Vt + (32 + lrow) * VT_PITCH + kb);
        v16h bv3 = *(const v16h*)(Vt + (48 + lrow) * VT_PITCH + kb);

#pragma unroll
        for (int qt = 0; qt < 2; ++qt) {
            v8f z = v8f_zero();
            v8f sL = __builtin_amdgcn_wmma_f32_16x16x32_f16(false, qa[qt][0], false, bkL0, (short)0, z, false, false);
            sL     = __builtin_amdgcn_wmma_f32_16x16x32_f16(false, qa[qt][1], false, bkL1, (short)0, sL, false, false);
            v8f sR = __builtin_amdgcn_wmma_f32_16x16x32_f16(false, qa[qt][0], false, bkR0, (short)0, z, false, false);
            sR     = __builtin_amdgcn_wmma_f32_16x16x32_f16(false, qa[qt][1], false, bkR1, (short)0, sR, false, false);

            // sigmoid (no max-trick needed: bounded (0,1)); accumulate denominator
#pragma unroll
            for (int i = 0; i < 8; ++i) {
                float pl = __builtin_amdgcn_rcpf(1.0f + __expf(-sL[i]));
                float pr = __builtin_amdgcn_rcpf(1.0f + __expf(-sR[i]));
                dsum[qt][i] += pl + pr;
                int m = i + (lhi << 3);                    // C-layout row
                Pw[m * 32 + lrow]      = (_Float16)pl;     // P row-major [m][k]
                Pw[m * 32 + 16 + lrow] = (_Float16)pr;
            }
            // reload P in A-layout (LDS in-order within wave)
            v16h ap = *(const v16h*)(Pw + lrow * 32 + lhi * 16);

            o[qt][0] = __builtin_amdgcn_wmma_f32_16x16x32_f16(false, ap, false, bv0, (short)0, o[qt][0], false, false);
            o[qt][1] = __builtin_amdgcn_wmma_f32_16x16x32_f16(false, ap, false, bv1, (short)0, o[qt][1], false, false);
            o[qt][2] = __builtin_amdgcn_wmma_f32_16x16x32_f16(false, ap, false, bv2, (short)0, o[qt][2], false, false);
            o[qt][3] = __builtin_amdgcn_wmma_f32_16x16x32_f16(false, ap, false, bv3, (short)0, o[qt][3], false, false);
        }
    }

    // ---- normalize, apply overlap-add blend weight, atomic-accumulate output ----
    const float istep = 1.0f / 255.0f;
#pragma unroll
    for (int qt = 0; qt < 2; ++qt) {
        // butterfly row-sum across the 16-lane halves (rows stay within a half)
#pragma unroll
        for (int m = 1; m < 16; m <<= 1) {
#pragma unroll
            for (int i = 0; i < 8; ++i)
                dsum[qt][i] += __shfl_xor(dsum[qt][i], m, 32);
        }
#pragma unroll
        for (int i = 0; i < 8; ++i) {
            int q = qwb + qt * 16 + i + (lhi << 3);        // row within window
            float wgt = (q < HALFW)
                          ? ((w == 0) ? 1.0f : (float)q * istep)
                          : (1.0f - (float)(q - HALFW) * istep);
            float f = wgt * __builtin_amdgcn_rcpf(dsum[qt][i]);
            float* orow = outB + (size_t)q * DHEAD + lrow;
            atomicAdd(orow +  0, o[qt][0][i] * f);
            atomicAdd(orow + 16, o[qt][1][i] * f);
            atomicAdd(orow + 32, o[qt][2][i] * f);
            atomicAdd(orow + 48, o[qt][3][i] * f);
        }
    }
}

extern "C" void kernel_launch(void* const* d_in, const int* in_sizes, int n_in,
                              void* d_out, int out_size, void* d_ws, size_t ws_size,
                              hipStream_t stream) {
    (void)in_sizes; (void)n_in; (void)d_ws; (void)ws_size;
    const float* Q     = (const float*)d_in[0];
    const float* K     = (const float*)d_in[1];
    const float* V     = (const float*)d_in[2];
    const float* scale = (const float*)d_in[3];
    float* out = (float*)d_out;

    // every output element receives exactly 1 or 2 weighted contributions via
    // atomicAdd starting from zero -> value-deterministic
    hipMemsetAsync(out, 0, (size_t)out_size * sizeof(float), stream);

    const size_t smem = (size_t)SMEM_HALFS * sizeof(_Float16);  // ~146 KB (<320 KB WGP LDS)
    hipFuncSetAttribute(reinterpret_cast<const void*>(swa_wmma_kernel),
                        hipFuncAttributeMaxDynamicSharedMemorySize, (int)smem);

    // one block per (b,h,window): 32 * 16 = 512 blocks, 16 waves each
    swa_wmma_kernel<<<dim3(32 * NWIN), dim3(THREADS), smem, stream>>>(Q, K, V, scale, out);
}
